// GeneGeneAttention_40175124087244
// MI455X (gfx1250) — matrix operand order repaired and checked by reading, not compiled, not run on hardware
//
#include <hip/hip_runtime.h>
#include <hip/hip_bf16.h>
#include <hip/hip_fp16.h>

typedef __attribute__((ext_vector_type(16))) _Float16 v16h;
typedef __attribute__((ext_vector_type(8)))  float    v8f;
typedef __attribute__((ext_vector_type(4)))  unsigned int u32x4;
typedef __attribute__((ext_vector_type(8)))  int          i32x8;
typedef __attribute__((ext_vector_type(4)))  int          i32x4;

#define D_MODEL  512
#define NHEAD    8
#define HEAD_DIM 64
#define BB       4
#define SS       2048
#define M_TOTAL  (BB*SS)      /* 8192 */
#define N_QKV    (3*D_MODEL)  /* 1536 */
#define MAX_GENES 8192

#if defined(__has_builtin)
#  if __has_builtin(__builtin_amdgcn_tensor_load_to_lds) && \
      __has_builtin(__builtin_amdgcn_s_wait_tensorcnt)
#    define USE_TDM 1
#  endif
#endif
#ifndef USE_TDM
#  define USE_TDM 0
#endif

// ---------------------------------------------------------------------------
// Fragment helpers (CDNA5 wave32 WMMA layouts, CDNA5 ISA §7.12.2)
// A (16x32 f16): lane m=lane&15 holds row m; kg=lane>>4 selects K groups
//   a[i] = K = kg*16 + i (contiguous, as B; not the ISA table order)
// B (32x16 f16): lane n=lane&15 holds col n; kg=lane>>4: b[i] = K = kg*16+i
// C (16x16 f32): element e: row = e + 8*kg, col = lane&15
// ---------------------------------------------------------------------------

static __device__ __forceinline__ v16h frag_a_h(const _Float16* __restrict__ p, int ld) {
    const int lane = threadIdx.x & 31;
    const int m  = lane & 15;
    const int kg = lane >> 4;
    v16h a;
#pragma unroll
    for (int i = 0; i < 16; ++i) a[i] = p[m * ld + kg * 16 + i];
    return a;
}

static __device__ __forceinline__ v16h frag_a_f(const float* __restrict__ p, int ld) {
    const int lane = threadIdx.x & 31;
    const int m  = lane & 15;
    const int kg = lane >> 4;
    v16h a;
#pragma unroll
    for (int i = 0; i < 16; ++i) a[i] = (_Float16)p[m * ld + kg * 16 + i];
    return a;
}

// B fragment, transposed source: B[k][n] = src[n][k]
static __device__ __forceinline__ v16h frag_bt_h(const _Float16* __restrict__ p, int ld) {
    const int lane = threadIdx.x & 31;
    const int n  = lane & 15;
    const int kg = lane >> 4;
    v16h b;
#pragma unroll
    for (int i = 0; i < 16; ++i) b[i] = p[n * ld + kg * 16 + i];
    return b;
}

static __device__ __forceinline__ v16h frag_bt_f(const float* __restrict__ p, int ld) {
    const int lane = threadIdx.x & 31;
    const int n  = lane & 15;
    const int kg = lane >> 4;
    v16h b;
#pragma unroll
    for (int i = 0; i < 16; ++i) b[i] = (_Float16)p[n * ld + kg * 16 + i];
    return b;
}

// B fragment, direct (row-major) source: B[k][n] = src[k][n]
static __device__ __forceinline__ v16h frag_b_h(const _Float16* __restrict__ p, int ld) {
    const int lane = threadIdx.x & 31;
    const int n  = lane & 15;
    const int kg = lane >> 4;
    v16h b;
#pragma unroll
    for (int i = 0; i < 16; ++i) b[i] = p[(kg * 16 + i) * ld + n];
    return b;
}

static __device__ __forceinline__ v8f wmma_f16(v16h a, v16h b, v8f c) {
    return __builtin_amdgcn_wmma_f32_16x16x32_f16(false, a, false, b, (short)0, c,
                                                  false, false);
}

#if USE_TDM
// ---------------------------------------------------------------------------
// TDM: async DMA of a 32x64 f16 tile (4 KB) from global to LDS.
// D# layout per CDNA5 ISA:
//  group0: [1:0] count=1 | [63:32] lds_addr | [120:64] global_addr | [127:126] type=2
//  group1: [17:16] data_size=1 (2B) | tensor_dim0=64 | tensor_dim1=32 |
//          tile_dim0=64 | tile_dim1=32 | tensor_dim0_stride=64
// Issued per-wave (EXEC ignored); tracked with TENSORcnt.
// ---------------------------------------------------------------------------
static __device__ __forceinline__ void tdm_load_tile_32x64(const _Float16* gsrc,
                                                           _Float16* ldst) {
    const unsigned long long ga = (unsigned long long)(uintptr_t)gsrc;
    const unsigned lds = (unsigned)(uintptr_t)ldst; // low 32 bits = LDS byte address

    u32x4 g0;
    g0[0] = 1u;                                      // count = 1 (valid descriptor)
    g0[1] = lds;                                     // lds_addr
    g0[2] = (unsigned)ga;                            // global_addr[31:0]
    g0[3] = (unsigned)(ga >> 32) | 0x80000000u;      // global_addr[56:32] | type=2

    i32x8 g1;
    g1[0] = 0x00010000;                              // data_size=2B, wg_mask=0
    g1[1] = (int)(64u << 16);                        // tensor_dim0 = 64 (bits 79:48 lo)
    g1[2] = (int)(32u << 16);                        // dim0 hi=0 | tensor_dim1=32 lo
    g1[3] = (int)(64u << 16);                        // dim1 hi=0 | tile_dim0=64
    g1[4] = 32;                                      // tile_dim1=32, tile_dim2=0
    g1[5] = 64;                                      // tensor_dim0_stride = 64 (lo 32)
    g1[6] = 0;                                       // stride0 hi | dim1_stride lo
    g1[7] = 0;

    const i32x4 z4 = {0, 0, 0, 0};
#if __clang_major__ >= 23
    const i32x8 z8 = {0, 0, 0, 0, 0, 0, 0, 0};
    __builtin_amdgcn_tensor_load_to_lds(g0, g1, z4, z4, z8, 0);
#else
    __builtin_amdgcn_tensor_load_to_lds(g0, g1, z4, z4, 0);
#endif
}
#endif // USE_TDM

// ---------------------------------------------------------------------------
// Kernel 1: QKV projection.  qkv[m][n] = sum_k X[m][k] * W[n][k] + bias[n]
// Output scattered into f16 workspace as q|k|v, each [B][H][S][64].
// One wave computes a 32x32 tile (2x2 WMMA tiles); operands are L2-resident.
// ---------------------------------------------------------------------------
__global__ __launch_bounds__(32)
void qkv_kernel(const float* __restrict__ X, const float* __restrict__ W,
                const float* __restrict__ bias, _Float16* __restrict__ qkv) {
    const int m0 = blockIdx.x * 32;
    const int n0 = blockIdx.y * 32;
    const int lane = threadIdx.x & 31;
    const int kg = lane >> 4;
    const int ln = lane & 15;

    v8f c00 = {}, c01 = {}, c10 = {}, c11 = {};
    for (int k0 = 0; k0 < D_MODEL; k0 += 32) {
        v16h a0 = frag_a_f(X + (size_t)m0 * D_MODEL + k0, D_MODEL);
        v16h a1 = frag_a_f(X + (size_t)(m0 + 16) * D_MODEL + k0, D_MODEL);
        v16h b0 = frag_bt_f(W + (size_t)n0 * D_MODEL + k0, D_MODEL);
        v16h b1 = frag_bt_f(W + (size_t)(n0 + 16) * D_MODEL + k0, D_MODEL);
        c00 = wmma_f16(a0, b0, c00);
        c01 = wmma_f16(a0, b1, c01);
        c10 = wmma_f16(a1, b0, c10);
        c11 = wmma_f16(a1, b1, c11);
    }

#pragma unroll
    for (int e = 0; e < 8; ++e) {
        const int rows[2] = { m0 + e + 8 * kg, m0 + 16 + e + 8 * kg };
        const int cols[2] = { n0 + ln, n0 + 16 + ln };
        const float vals[2][2] = { { c00[e], c01[e] }, { c10[e], c11[e] } };
#pragma unroll
        for (int im = 0; im < 2; ++im) {
#pragma unroll
            for (int in = 0; in < 2; ++in) {
                const int m = rows[im];
                const int n = cols[in];
                const int sec = n >> 9;          // 0=q 1=k 2=v
                const int nn  = n & (D_MODEL - 1);
                const int h   = nn >> 6;
                const int d   = nn & (HEAD_DIM - 1);
                const int b   = m >> 11;
                const int s   = m & (SS - 1);
                const size_t idx = (size_t)sec * ((size_t)BB * NHEAD * SS * HEAD_DIM) +
                                   (((size_t)(b * NHEAD + h) * SS + s) * HEAD_DIM + d);
                qkv[idx] = (_Float16)(vals[im][in] + bias[n]);
            }
        }
    }
}

// ---------------------------------------------------------------------------
// Kernel 2: flash attention, 4 waves/block (64 queries), fused gene-bias gather.
// K/V 32-key tiles staged into LDS once per block via TDM (double-buffered),
// shared by all 4 waves.
// ---------------------------------------------------------------------------
__global__ __launch_bounds__(128)
void attn_kernel(const _Float16* __restrict__ qkv, const int* __restrict__ gidx,
                 const float* __restrict__ gbias, _Float16* __restrict__ aout) {
    const int wave = threadIdx.x >> 5;
    const int lane = threadIdx.x & 31;
    const int kg = lane >> 4;
    const int ln = lane & 15;
    const int h  = blockIdx.y;   // 0..7
    const int b  = blockIdx.z;   // 0..3
    const int qt = blockIdx.x * 4 + wave;  // 16-query tile index, 0..127
    const int q0 = qt * 16;

    const size_t plane = (size_t)(b * NHEAD + h) * SS * HEAD_DIM;
    const size_t qkv_stride = (size_t)BB * NHEAD * SS * HEAD_DIM;
    const _Float16* Q = qkv + plane;
    const _Float16* K = qkv + qkv_stride + plane;
    const _Float16* V = qkv + 2 * qkv_stride + plane;

    __shared__ _Float16 kbuf[2][32 * HEAD_DIM];
    __shared__ _Float16 vbuf[2][32 * HEAD_DIM];
    __shared__ _Float16 ptile[4][16 * 32];
    _Float16* pt = ptile[wave];

    // Q A-fragments for head-dim chunks [0,32) and [32,64)
    const v16h qa0 = frag_a_h(Q + (size_t)q0 * HEAD_DIM + 0, HEAD_DIM);
    const v16h qa1 = frag_a_h(Q + (size_t)q0 * HEAD_DIM + 32, HEAD_DIM);

    // gene row indices for the 8 query rows this lane owns (C layout rows)
    int qg[8];
#pragma unroll
    for (int e = 0; e < 8; ++e)
        qg[e] = gidx[b * SS + q0 + e + 8 * kg] * MAX_GENES;

    float m_i[8], l_i[8];
#pragma unroll
    for (int e = 0; e < 8; ++e) { m_i[e] = -3.0e38f; l_i[e] = 0.0f; }
    v8f o0 = {}, o1 = {}, o2 = {}, o3 = {};

    const float scale = 0.125f; // 1/sqrt(64)
    const int NBLK = SS / 32;

#if USE_TDM
    if (threadIdx.x < 32) {                       // wave 0 issues TDM prefetch
        tdm_load_tile_32x64(K, kbuf[0]);
        tdm_load_tile_32x64(V, vbuf[0]);
    }
#endif

    for (int it = 0; it < NBLK; ++it) {
        const int k0 = it * 32;
#if USE_TDM
        const int cur = it & 1;
        if (threadIdx.x < 32) {
            if (it + 1 < NBLK) {                  // prefetch next tile pair
                tdm_load_tile_32x64(K + (size_t)(k0 + 32) * HEAD_DIM, kbuf[cur ^ 1]);
                tdm_load_tile_32x64(V + (size_t)(k0 + 32) * HEAD_DIM, vbuf[cur ^ 1]);
                __builtin_amdgcn_s_wait_tensorcnt((short)2);  // current pair done
            } else {
                __builtin_amdgcn_s_wait_tensorcnt((short)0);
            }
        }
#else
        const int cur = 0;
        {   // cooperative staged copy: 32*64 halfs = 256 x int4
            const int4* ks = (const int4*)(K + (size_t)k0 * HEAD_DIM);
            const int4* vs = (const int4*)(V + (size_t)k0 * HEAD_DIM);
            int4* kd = (int4*)kbuf[0];
            int4* vd = (int4*)vbuf[0];
            for (int i = threadIdx.x; i < 32 * HEAD_DIM / 8; i += 128) {
                kd[i] = ks[i];
                vd[i] = vs[i];
            }
        }
#endif
        __syncthreads();  // tile visible to all waves

        const _Float16* Kc = kbuf[cur];
        const _Float16* Vc = vbuf[cur];

        // ---- scores: 16 queries x 32 keys (two 16x16 tiles, K=64) ----
        v8f s0 = {}, s1 = {};
        {
            v16h kb;
            kb = frag_bt_h(Kc + 0 * HEAD_DIM + 0,   HEAD_DIM); s0 = wmma_f16(qa0, kb, s0);
            kb = frag_bt_h(Kc + 0 * HEAD_DIM + 32,  HEAD_DIM); s0 = wmma_f16(qa1, kb, s0);
            kb = frag_bt_h(Kc + 16 * HEAD_DIM + 0,  HEAD_DIM); s1 = wmma_f16(qa0, kb, s1);
            kb = frag_bt_h(Kc + 16 * HEAD_DIM + 32, HEAD_DIM); s1 = wmma_f16(qa1, kb, s1);
        }

        // ---- fused gene-bias gather (the HBM-bound part) ----
        const int kgi0 = gidx[b * SS + k0 + ln];
        const int kgi1 = gidx[b * SS + k0 + 16 + ln];
#pragma unroll
        for (int e = 0; e < 8; ++e) {
            s0[e] = s0[e] * scale + gbias[(size_t)(qg[e] + kgi0)];
            s1[e] = s1[e] * scale + gbias[(size_t)(qg[e] + kgi1)];
        }

        // ---- online softmax (row reductions across 16-lane halves) ----
#pragma unroll
        for (int e = 0; e < 8; ++e) {
            float mx = fmaxf(s0[e], s1[e]);
#pragma unroll
            for (int w = 1; w < 16; w <<= 1)
                mx = fmaxf(mx, __shfl_xor(mx, w, 32));
            const float mnew = fmaxf(m_i[e], mx);
            const float p0 = __expf(s0[e] - mnew);
            const float p1 = __expf(s1[e] - mnew);
            float rs = p0 + p1;
#pragma unroll
            for (int w = 1; w < 16; w <<= 1)
                rs += __shfl_xor(rs, w, 32);
            const float alpha = __expf(m_i[e] - mnew);
            l_i[e] = l_i[e] * alpha + rs;
            m_i[e] = mnew;
            s0[e] = p0;
            s1[e] = p1;
            o0[e] *= alpha; o1[e] *= alpha; o2[e] *= alpha; o3[e] *= alpha;
        }

        // ---- re-layout P: C fragment -> per-wave LDS tile -> A fragment ----
        // wave-local exchange: LDS ops from one wave are in-order, no barrier
#pragma unroll
        for (int e = 0; e < 8; ++e) {
            pt[(e + 8 * kg) * 32 + ln]      = (_Float16)s0[e];
            pt[(e + 8 * kg) * 32 + 16 + ln] = (_Float16)s1[e];
        }
        __builtin_amdgcn_wave_barrier();
        const v16h pa = frag_a_h(pt, 32);

        // ---- O += P @ V (16x64, four 16x16 tiles, K=32) ----
        v16h vb;
        vb = frag_b_h(Vc + 0,  HEAD_DIM); o0 = wmma_f16(pa, vb, o0);
        vb = frag_b_h(Vc + 16, HEAD_DIM); o1 = wmma_f16(pa, vb, o1);
        vb = frag_b_h(Vc + 32, HEAD_DIM); o2 = wmma_f16(pa, vb, o2);
        vb = frag_b_h(Vc + 48, HEAD_DIM); o3 = wmma_f16(pa, vb, o3);

        __syncthreads();  // all waves done with this tile before it is overwritten
    }

    // ---- finalize: O /= l, write f16 to [B][S][H][64] (row-major 8192x512) ----
#pragma unroll
    for (int e = 0; e < 8; ++e) {
        const float il = 1.0f / l_i[e];
        const int s = q0 + e + 8 * kg;
        const size_t base = (((size_t)b * SS + s) * NHEAD + h) * HEAD_DIM;
        aout[base + 0  + ln] = (_Float16)(o0[e] * il);
        aout[base + 16 + ln] = (_Float16)(o1[e] * il);
        aout[base + 32 + ln] = (_Float16)(o2[e] * il);
        aout[base + 48 + ln] = (_Float16)(o3[e] * il);
    }
}

// ---------------------------------------------------------------------------
// Kernel 3: output projection. out[m][n] = sum_k A[m][k]*W[n][k] + b[n], f32 out
// ---------------------------------------------------------------------------
__global__ __launch_bounds__(32)
void outproj_kernel(const _Float16* __restrict__ A, const float* __restrict__ W,
                    const float* __restrict__ bias, float* __restrict__ out) {
    const int m0 = blockIdx.x * 32;
    const int n0 = blockIdx.y * 32;
    const int lane = threadIdx.x & 31;
    const int kg = lane >> 4;
    const int ln = lane & 15;

    v8f c00 = {}, c01 = {}, c10 = {}, c11 = {};
    for (int k0 = 0; k0 < D_MODEL; k0 += 32) {
        v16h a0 = frag_a_h(A + (size_t)m0 * D_MODEL + k0, D_MODEL);
        v16h a1 = frag_a_h(A + (size_t)(m0 + 16) * D_MODEL + k0, D_MODEL);
        v16h b0 = frag_bt_f(W + (size_t)n0 * D_MODEL + k0, D_MODEL);
        v16h b1 = frag_bt_f(W + (size_t)(n0 + 16) * D_MODEL + k0, D_MODEL);
        c00 = wmma_f16(a0, b0, c00);
        c01 = wmma_f16(a0, b1, c01);
        c10 = wmma_f16(a1, b0, c10);
        c11 = wmma_f16(a1, b1, c11);
    }

#pragma unroll
    for (int e = 0; e < 8; ++e) {
        const int r0 = m0 + e + 8 * kg;
        const int r1 = m0 + 16 + e + 8 * kg;
        out[(size_t)r0 * D_MODEL + n0 + ln]      = c00[e] + bias[n0 + ln];
        out[(size_t)r0 * D_MODEL + n0 + 16 + ln] = c01[e] + bias[n0 + 16 + ln];
        out[(size_t)r1 * D_MODEL + n0 + ln]      = c10[e] + bias[n0 + ln];
        out[(size_t)r1 * D_MODEL + n0 + 16 + ln] = c11[e] + bias[n0 + 16 + ln];
    }
}

// ---------------------------------------------------------------------------
extern "C" void kernel_launch(void* const* d_in, const int* in_sizes, int n_in,
                              void* d_out, int out_size, void* d_ws, size_t ws_size,
                              hipStream_t stream) {
    (void)in_sizes; (void)n_in; (void)out_size; (void)ws_size;

    const float* query  = (const float*)d_in[0];
    const int*   gidx   = (const int*)  d_in[1];
    const float* w_in   = (const float*)d_in[2];
    const float* b_in   = (const float*)d_in[3];
    const float* w_out  = (const float*)d_in[4];
    const float* b_out  = (const float*)d_in[5];
    const float* gbias  = (const float*)d_in[6];

    _Float16* qkv  = (_Float16*)d_ws;                               // 24 MB
    _Float16* aout = qkv + (size_t)3 * M_TOTAL * D_MODEL;           // +8 MB
    float*    out  = (float*)d_out;

    qkv_kernel<<<dim3(M_TOTAL / 32, N_QKV / 32), 32, 0, stream>>>(query, w_in, b_in, qkv);
    attn_kernel<<<dim3(SS / 16 / 4, NHEAD, BB), 128, 0, stream>>>(qkv, gidx, gbias, aout);
    outproj_kernel<<<dim3(M_TOTAL / 32, D_MODEL / 32), 32, 0, stream>>>(aout, w_out, b_out, out);
}
